// RSSM_738734375714
// MI455X (gfx1250) — matrix operand order repaired and checked by reading, not compiled, not run on hardware
//
#include <hip/hip_runtime.h>
#include <hip/hip_bf16.h>

// ---------------- problem constants ----------------
#define BB 32
#define LL 64
#define DETERN 4096
#define STOCHN 32
#define DISCN 32
#define HIDDEN 1024
#define NBLOCKS 8
#define EMBEDN 4096
#define ACTDIM 6
#define FLATSTOCH 1024      // 32*32
#define DPB 512             // DETER / BLOCKS
#define INPB 3584           // 3*HIDDEN + DPB
#define BT (BB*LL)          // 2048

typedef __bf16 bf16;
typedef __attribute__((ext_vector_type(16))) __bf16 bf16x16;
typedef __attribute__((ext_vector_type(8)))  __bf16 bf16x8;
typedef __attribute__((ext_vector_type(8)))  float  f32x8;

// A-fragment (16x32 bf16, one row per lane): p = &A[row*lda + k0 + ((lane>>4)<<4)].
// Elements 0..15 = K[kb..kb+15].
__device__ __forceinline__ bf16x16 load_a_chunk(const bf16* p) {
    union { bf16x16 v; bf16x8 h[2]; } u;
    u.h[0] = *reinterpret_cast<const bf16x8*>(p);
    u.h[1] = *reinterpret_cast<const bf16x8*>(p + 8);
    return u.v;
}

#define WMMA_BF16(a, b, c) \
    __builtin_amdgcn_wmma_f32_16x16x32_bf16(false, (a), false, (b), (short)0, (c), false, false)

// ---------------------------------------------------------------------------
// M=32 GEMM: C[32,N](f32) = A[32,K](bf16) * W[N,K]^T (bf16, pre-transposed).
// One wave = full 32-row stripe of one 16-col tile: B fragment loaded once,
// two A fragments, two WMMAs per 32-K step.  Optional block-diagonal A column
// offset (grouped GRU matmul): acol = (n0 / nb_group) * a_group_stride.
// ---------------------------------------------------------------------------
__global__ __launch_bounds__(256)
void k_gemm32(const bf16* __restrict__ A, int lda,
              const bf16* __restrict__ W, int ldb,
              float* __restrict__ C, int ldc,
              int N, int K,
              const float* __restrict__ bias,
              int nb_group, int a_group_stride)
{
    const int lane = threadIdx.x & 31;
    const int wave = threadIdx.x >> 5;
    const int tile = blockIdx.x * 8 + wave;
    if (tile >= (N >> 4)) return;
    const int n0 = tile << 4;
    const int acol = (n0 / nb_group) * a_group_stride;

    const int r = lane & 15;
    const int half8 = (lane >> 4) << 3;
    const bf16* arow0 = A + (size_t)r * lda + acol + ((lane >> 4) << 4);
    const bf16* arow1 = arow0 + (size_t)16 * lda;
    const bf16* brow  = W + (size_t)(n0 + r) * ldb + ((lane >> 4) << 4);

    f32x8 acc0 = {}, acc1 = {};
    for (int k0 = 0; k0 < K; k0 += 32) {
        bf16x16 b  = *reinterpret_cast<const bf16x16*>(brow + k0);
        bf16x16 a0 = load_a_chunk(arow0 + k0);
        bf16x16 a1 = load_a_chunk(arow1 + k0);
        acc0 = WMMA_BF16(a0, b, acc0);
        acc1 = WMMA_BF16(a1, b, acc1);
    }

    const int rbase = half8;
    const int ccol  = n0 + r;
    const float badd = bias ? bias[ccol] : 0.0f;
#pragma unroll
    for (int i = 0; i < 8; ++i) {
        C[(size_t)(rbase + i)      * ldc + ccol] = acc0[i] + badd;
        C[(size_t)(rbase + 16 + i) * ldc + ccol] = acc1[i] + badd;
    }
}

// ---------------------------------------------------------------------------
// Block-diagonal hidden matmul (M=32, K=3584 per block, N=4096 total):
//   h[b, g*512+n] = sum_k [ deterM[b,g*512..] | x[b,0..3071] ][k] * w_hid[g,k,n]
// deterM is already reset-masked.  Same 2-row-tile scheme as k_gemm32.
// ---------------------------------------------------------------------------
__global__ __launch_bounds__(256)
void k_gemm_hid32(const bf16* __restrict__ deterM, // [32,4096] (masked)
                  const bf16* __restrict__ x,      // [32,3072]
                  const bf16* __restrict__ Wt,     // [4096][3584]
                  float* __restrict__ C)           // [32,4096]
{
    const int lane = threadIdx.x & 31;
    const int wave = threadIdx.x >> 5;
    const int tile = blockIdx.x * 8 + wave;        // 256 tiles
    if (tile >= 256) return;
    const int n0 = tile << 4;
    const int g  = n0 >> 9;

    const int r = lane & 15;
    const int half8 = (lane >> 4) << 3;
    const bf16* ad0 = deterM + (size_t)r * DETERN + (g << 9) + ((lane >> 4) << 4);
    const bf16* ad1 = ad0 + (size_t)16 * DETERN;
    const bf16* ax0 = x + (size_t)r * (3 * HIDDEN) + ((lane >> 4) << 4);
    const bf16* ax1 = ax0 + (size_t)16 * (3 * HIDDEN);
    const bf16* brow = Wt + (size_t)(n0 + r) * INPB + ((lane >> 4) << 4);

    f32x8 acc0 = {}, acc1 = {};
    for (int k0 = 0; k0 < DPB; k0 += 32) {
        bf16x16 b  = *reinterpret_cast<const bf16x16*>(brow + k0);
        bf16x16 a0 = load_a_chunk(ad0 + k0);
        bf16x16 a1 = load_a_chunk(ad1 + k0);
        acc0 = WMMA_BF16(a0, b, acc0);
        acc1 = WMMA_BF16(a1, b, acc1);
    }
    for (int k0 = 0; k0 < 3 * HIDDEN; k0 += 32) {
        bf16x16 b  = *reinterpret_cast<const bf16x16*>(brow + DPB + k0);
        bf16x16 a0 = load_a_chunk(ax0 + k0);
        bf16x16 a1 = load_a_chunk(ax1 + k0);
        acc0 = WMMA_BF16(a0, b, acc0);
        acc1 = WMMA_BF16(a1, b, acc1);
    }

    const int ccol = n0 + r;
#pragma unroll
    for (int i = 0; i < 8; ++i) {
        C[(size_t)(half8 + i)      * DETERN + ccol] = acc0[i];
        C[(size_t)(half8 + 16 + i) * DETERN + ccol] = acc1[i];
    }
}

// ---------------------------------------------------------------------------
// Big-M GEMM (posterior path, M=2048): one wave = 1 m-tile x 4 n-tiles.
// A fragment loaded once per 4 WMMAs.
// ---------------------------------------------------------------------------
__global__ __launch_bounds__(256)
void k_gemm_big(const bf16* __restrict__ A, int lda,
                const bf16* __restrict__ W, int ldb,
                float* __restrict__ C, int ldc,
                int M, int N, int K,
                const float* __restrict__ bias)
{
    const int lane = threadIdx.x & 31;
    const int wave = threadIdx.x >> 5;
    const int nq = N >> 6;                    // groups of 4 n-tiles
    const int ntiles = (M >> 4) * nq;
    const int tile = blockIdx.x * 8 + wave;
    if (tile >= ntiles) return;
    const int n0 = (tile % nq) << 6;
    const int m0 = (tile / nq) << 4;

    const int r = lane & 15;
    const int half8 = (lane >> 4) << 3;
    const bf16* arow = A + (size_t)(m0 + r) * lda + ((lane >> 4) << 4);
    const bf16* b0 = W + (size_t)(n0 + r) * ldb + ((lane >> 4) << 4);
    const bf16* b1 = b0 + (size_t)16 * ldb;
    const bf16* b2 = b0 + (size_t)32 * ldb;
    const bf16* b3 = b0 + (size_t)48 * ldb;

    f32x8 c0 = {}, c1 = {}, c2 = {}, c3 = {};
    for (int k0 = 0; k0 < K; k0 += 32) {
        bf16x16 a  = load_a_chunk(arow + k0);
        bf16x16 w0 = *reinterpret_cast<const bf16x16*>(b0 + k0);
        bf16x16 w1 = *reinterpret_cast<const bf16x16*>(b1 + k0);
        bf16x16 w2 = *reinterpret_cast<const bf16x16*>(b2 + k0);
        bf16x16 w3 = *reinterpret_cast<const bf16x16*>(b3 + k0);
        c0 = WMMA_BF16(a, w0, c0);
        c1 = WMMA_BF16(a, w1, c1);
        c2 = WMMA_BF16(a, w2, c2);
        c3 = WMMA_BF16(a, w3, c3);
    }

    const int crow = m0 + half8;
#pragma unroll
    for (int j = 0; j < 4; ++j) {
        f32x8 acc = (j == 0) ? c0 : (j == 1) ? c1 : (j == 2) ? c2 : c3;
        const int ccol = n0 + j * 16 + r;
        const float badd = bias ? bias[ccol] : 0.0f;
#pragma unroll
        for (int i = 0; i < 8; ++i)
            C[(size_t)(crow + i) * ldc + ccol] = acc[i] + badd;
    }
}

// ---------------------------------------------------------------------------
// RMSNorm(+bias) + SiLU, output bf16.  One block per row.
// ---------------------------------------------------------------------------
__global__ __launch_bounds__(256)
void k_rmssilu(const float* __restrict__ raw, int ldraw,
               bf16* __restrict__ out, int ldout,
               const float* __restrict__ bias, const float* __restrict__ gain,
               int W)
{
    const int row = blockIdx.x;
    __shared__ float red[256];
    const float* rp = raw + (size_t)row * ldraw;
    float s = 0.0f;
    for (int j = threadIdx.x; j < W; j += 256) {
        float v = rp[j] + bias[j];
        s += v * v;
    }
    red[threadIdx.x] = s;
    __syncthreads();
    for (int off = 128; off > 0; off >>= 1) {
        if (threadIdx.x < off) red[threadIdx.x] += red[threadIdx.x + off];
        __syncthreads();
    }
    const float inv = rsqrtf(red[0] / (float)W + 1e-4f);
    bf16* op = out + (size_t)row * ldout;
    for (int j = threadIdx.x; j < W; j += 256) {
        float v = (rp[j] + bias[j]) * inv * gain[j];
        op[j] = (bf16)(v / (1.0f + expf(-v)));
    }
}

// Fused over the three 1024-wide input-projection segments (x0|x1|x2).
__global__ __launch_bounds__(256)
void k_rmssilu_x(const float* __restrict__ raw,      // [32,3072]
                 bf16* __restrict__ out,             // [32,3072]
                 const float* __restrict__ b0, const float* __restrict__ g0,
                 const float* __restrict__ b1, const float* __restrict__ g1,
                 const float* __restrict__ b2, const float* __restrict__ g2)
{
    const int row = blockIdx.x;
    const int seg = blockIdx.y;
    const float* bias = (seg == 0) ? b0 : (seg == 1) ? b1 : b2;
    const float* gain = (seg == 0) ? g0 : (seg == 1) ? g1 : g2;
    __shared__ float red[256];
    const float* rp = raw + (size_t)row * (3 * HIDDEN) + seg * HIDDEN;
    float s = 0.0f;
    for (int j = threadIdx.x; j < HIDDEN; j += 256) {
        float v = rp[j] + bias[j];
        s += v * v;
    }
    red[threadIdx.x] = s;
    __syncthreads();
    for (int off = 128; off > 0; off >>= 1) {
        if (threadIdx.x < off) red[threadIdx.x] += red[threadIdx.x + off];
        __syncthreads();
    }
    const float inv = rsqrtf(red[0] / (float)HIDDEN + 1e-4f);
    bf16* op = out + (size_t)row * (3 * HIDDEN) + seg * HIDDEN;
    for (int j = threadIdx.x; j < HIDDEN; j += 256) {
        float v = (rp[j] + bias[j]) * inv * gain[j];
        op[j] = (bf16)(v / (1.0f + expf(-v)));
    }
}

// ---------------------------------------------------------------------------
// Action projection (K=6, plain VALU): xraw[:,2048:3072] = norm(mask(a)) @ w_in2
// ---------------------------------------------------------------------------
__global__ __launch_bounds__(256)
void k_act(const float* __restrict__ action,   // [B,L,6]
           const int* __restrict__ reset,      // [B,L]
           const float* __restrict__ w_in2,    // [6,1024]
           float* __restrict__ out,            // xraw + 2048, row stride 3072
           int t)
{
    const int idx = blockIdx.x * 256 + threadIdx.x;   // 32*1024
    const int b = idx >> 10, j = idx & 1023;
    const bool kill = reset[b * LL + t] != 0;
    float s = 0.0f;
#pragma unroll
    for (int i = 0; i < ACTDIM; ++i) {
        float a = kill ? 0.0f : action[((size_t)b * LL + t) * ACTDIM + i];
        float d = fabsf(a); d = d > 1.0f ? d : 1.0f;
        s += (a / d) * w_in2[i * HIDDEN + j];
    }
    out[(size_t)b * (3 * HIDDEN) + j] = s;
}

// ---------------------------------------------------------------------------
// Softmax + unimix + Gumbel argmax + straight-through, per (b,t,s).
// Writes f32 stochs output and a reset-premasked bf16 copy for step t+1.
// ---------------------------------------------------------------------------
__global__ __launch_bounds__(256)
void k_sample(const float* __restrict__ logits,   // [2048,1024] (d_out region)
              const float* __restrict__ unif,     // [2048,1024]
              const int* __restrict__ reset,      // [B,L]
              float* __restrict__ stoch_out,      // d_out stochs [2048,1024]
              bf16* __restrict__ stoch_bf)        // [65][32][1024]
{
    const int idx = blockIdx.x * 256 + threadIdx.x;   // 2048*32
    const int bt = idx >> 5, s = idx & 31;
    const int b = bt / LL, t = bt % LL;
    const float* lp = logits + (size_t)bt * FLATSTOCH + s * DISCN;
    const float* up = unif   + (size_t)bt * FLATSTOCH + s * DISCN;

    float l[DISCN];
    float m = -3.0e38f;
#pragma unroll
    for (int i = 0; i < DISCN; ++i) { l[i] = lp[i]; m = fmaxf(m, l[i]); }
    float sum = 0.0f;
#pragma unroll
    for (int i = 0; i < DISCN; ++i) { l[i] = expf(l[i] - m); sum += l[i]; }
    const float inv = 1.0f / sum;

    float best = -3.0e38f; int bi = 0;
#pragma unroll
    for (int i = 0; i < DISCN; ++i) {
        float p = l[i] * inv * 0.99f + (0.01f / 32.0f);
        l[i] = p;
        float sc = logf(p) + (-logf(-logf(up[i])));
        if (sc > best) { best = sc; bi = i; }
    }

    // reset mask for the step that will consume this stoch (t+1)
    const bool killn = (t + 1 < LL) ? (reset[b * LL + t + 1] != 0) : false;
#pragma unroll
    for (int i = 0; i < DISCN; ++i) {
        float oh  = (i == bi) ? 1.0f : 0.0f;
        float val = (oh + l[i]) - l[i];   // straight-through forward value
        stoch_out[(size_t)bt * FLATSTOCH + s * DISCN + i] = val;
        stoch_bf[((size_t)(t + 1) * BB + b) * FLATSTOCH + s * DISCN + i]
            = killn ? (bf16)0.0f : (bf16)val;
    }
}

// ---------------------------------------------------------------------------
// GRU gate update.  Also emits the reset-premasked bf16 deter for step t+1.
// ---------------------------------------------------------------------------
__global__ __launch_bounds__(256)
void k_gru_update(const float* __restrict__ g3,      // [32,12288]
                  const float* __restrict__ b_gru,   // [12288]
                  const int* __restrict__ reset,     // [B,L]
                  float* __restrict__ deter_f32,     // [32,4096] state (unmasked)
                  bf16*  __restrict__ deter_mbf,     // [32,4096] masked for t+1
                  float* __restrict__ out_deters,    // d_out deters region
                  int t)
{
    const int idx = blockIdx.x * 256 + threadIdx.x;  // 32*4096
    const int b = idx >> 12, d = idx & 4095;
    const int g = d >> 9, j = d & 511;
    const size_t base = (size_t)b * (3 * DETERN) + g * (3 * DPB);
    const int   gb   = g * (3 * DPB);
    float rv = g3[base + j]           + b_gru[gb + j];
    float cv = g3[base + DPB + j]     + b_gru[gb + DPB + j];
    float uv = g3[base + 2 * DPB + j] + b_gru[gb + 2 * DPB + j] - 1.0f;
    float r    = 1.0f / (1.0f + expf(-rv));
    float cand = tanhf(r * cv);
    float u    = 1.0f / (1.0f + expf(-uv));
    float dprev = (reset[b * LL + t] != 0) ? 0.0f : deter_f32[idx];
    float dn = u * cand + (1.0f - u) * dprev;
    deter_f32[idx] = dn;
    const bool killn = (t + 1 < LL) ? (reset[b * LL + t + 1] != 0) : false;
    deter_mbf[idx] = killn ? (bf16)0.0f : (bf16)dn;
    out_deters[((size_t)b * LL + t) * DETERN + d] = dn;
}

// ---------------------------------------------------------------------------
// Weight transpose + f32->bf16:  src [Z,K,N] f32  ->  dst [Z,N,K] bf16
// ---------------------------------------------------------------------------
__global__ __launch_bounds__(256)
void k_transpose(const float* __restrict__ src, bf16* __restrict__ dst,
                 int K, int N)
{
    __shared__ float tile[32][33];
    const int z = blockIdx.z;
    src += (size_t)z * K * N;
    dst += (size_t)z * N * K;
    const int k0 = blockIdx.x * 32, n0 = blockIdx.y * 32;
    const int tx = threadIdx.x & 31, ty = threadIdx.x >> 5;   // ty 0..7
    for (int i = ty; i < 32; i += 8) {
        int k = k0 + i, n = n0 + tx;
        tile[i][tx] = (k < K && n < N) ? src[(size_t)k * N + n] : 0.0f;
    }
    __syncthreads();
    for (int i = ty; i < 32; i += 8) {
        int n = n0 + i, k = k0 + tx;
        if (n < N && k < K) dst[(size_t)n * K + k] = (bf16)tile[tx][i];
    }
}

__global__ __launch_bounds__(256)
void k_cvt_bf16(const float* __restrict__ src, bf16* __restrict__ dst, size_t n)
{
    size_t i = (size_t)blockIdx.x * 256 + threadIdx.x;
    if (i < n) dst[i] = (bf16)src[i];
}

// ---------------------------------------------------------------------------
extern "C" void kernel_launch(void* const* d_in, const int* in_sizes, int n_in,
                              void* d_out, int out_size, void* d_ws, size_t ws_size,
                              hipStream_t stream)
{
    (void)in_sizes; (void)n_in; (void)out_size; (void)ws_size;

    const float* embed   = (const float*)d_in[0];
    const float* action  = (const float*)d_in[1];
    const int*   reset   = (const int*)  d_in[2];
    const float* unif    = (const float*)d_in[3];
    const float* w_in0   = (const float*)d_in[4];
    const float* b_in0   = (const float*)d_in[5];
    const float* g_in0   = (const float*)d_in[6];
    const float* w_in1   = (const float*)d_in[7];
    const float* b_in1   = (const float*)d_in[8];
    const float* g_in1   = (const float*)d_in[9];
    const float* w_in2   = (const float*)d_in[10];
    const float* b_in2   = (const float*)d_in[11];
    const float* g_in2   = (const float*)d_in[12];
    const float* w_hid   = (const float*)d_in[13];
    const float* b_hid   = (const float*)d_in[14];
    const float* g_hid   = (const float*)d_in[15];
    const float* w_gru   = (const float*)d_in[16];
    const float* b_gru   = (const float*)d_in[17];
    const float* w_obs   = (const float*)d_in[18];
    const float* b_obs   = (const float*)d_in[19];
    const float* g_obs   = (const float*)d_in[20];
    const float* w_logit = (const float*)d_in[21];
    const float* b_logit = (const float*)d_in[22];

    float* out_stochs = (float*)d_out;                                 // [B,L,32,32]
    float* out_deters = (float*)d_out + (size_t)BT * FLATSTOCH;        // [B,L,4096]
    float* out_logits = out_deters + (size_t)BT * DETERN;              // [B,L,32,32]

    // ---- workspace carve-up (256B aligned) ----
    char* ws = (char*)d_ws;
    size_t off = 0;
    auto take = [&](size_t bytes) -> char* {
        char* p = ws + off;
        off = (off + bytes + 255) & ~(size_t)255;
        return p;
    };
    bf16*  w_in0_t   = (bf16*) take((size_t)HIDDEN * DETERN * 2);
    bf16*  w_in1_t   = (bf16*) take((size_t)HIDDEN * FLATSTOCH * 2);
    bf16*  w_obs_t   = (bf16*) take((size_t)HIDDEN * EMBEDN * 2);
    bf16*  w_logit_t = (bf16*) take((size_t)FLATSTOCH * HIDDEN * 2);
    bf16*  w_hid_t   = (bf16*) take((size_t)NBLOCKS * DPB * INPB * 2);
    bf16*  w_gru_t   = (bf16*) take((size_t)NBLOCKS * 3 * DPB * DPB * 2);
    bf16*  e_bf      = (bf16*) take((size_t)BT * EMBEDN * 2);
    float* hobs_raw  = (float*)take((size_t)BT * HIDDEN * 4);
    bf16*  hobs_bf   = (bf16*) take((size_t)BT * HIDDEN * 2);
    bf16*  stoch_bf  = (bf16*) take((size_t)(LL + 1) * BB * FLATSTOCH * 2);
    float* deter_f32 = (float*)take((size_t)BB * DETERN * 4);
    bf16*  deter_mbf = (bf16*) take((size_t)BB * DETERN * 2);
    float* xraw      = (float*)take((size_t)BB * 3 * HIDDEN * 4);
    bf16*  x_bf      = (bf16*) take((size_t)BB * 3 * HIDDEN * 2);
    float* h_raw     = (float*)take((size_t)BB * DETERN * 4);
    bf16*  h_bf      = (bf16*) take((size_t)BB * DETERN * 2);
    float* g3        = (float*)take((size_t)BB * 3 * DETERN * 4);

    const dim3 tb(256);

    // ---- Phase 0: weight transposition to bf16 [N,K] ----
    k_transpose<<<dim3(DETERN/32, HIDDEN/32, 1), tb, 0, stream>>>(w_in0,   w_in0_t,   DETERN, HIDDEN);
    k_transpose<<<dim3(FLATSTOCH/32, HIDDEN/32, 1), tb, 0, stream>>>(w_in1, w_in1_t,  FLATSTOCH, HIDDEN);
    k_transpose<<<dim3(EMBEDN/32, HIDDEN/32, 1), tb, 0, stream>>>(w_obs,   w_obs_t,   EMBEDN, HIDDEN);
    k_transpose<<<dim3(HIDDEN/32, FLATSTOCH/32, 1), tb, 0, stream>>>(w_logit, w_logit_t, HIDDEN, FLATSTOCH);
    k_transpose<<<dim3(INPB/32, DPB/32, NBLOCKS), tb, 0, stream>>>(w_hid,   w_hid_t,   INPB, DPB);
    k_transpose<<<dim3(DPB/32, (3*DPB)/32, NBLOCKS), tb, 0, stream>>>(w_gru, w_gru_t,  DPB, 3*DPB);

    {   // embed -> bf16
        size_t n = (size_t)BT * EMBEDN;
        k_cvt_bf16<<<(unsigned)((n + 255) / 256), tb, 0, stream>>>(embed, e_bf, n);
    }
    // zero recurrent state and stoch slot 0 (zero is reset-invariant)
    hipMemsetAsync(deter_f32, 0, (size_t)BB * DETERN * 4, stream);
    hipMemsetAsync(deter_mbf, 0, (size_t)BB * DETERN * 2, stream);
    hipMemsetAsync(stoch_bf,  0, (size_t)BB * FLATSTOCH * 2, stream);

    // ---- Phase 1: posterior path for all (b,t) in parallel ----
    k_gemm_big<<<(BT/16)*(HIDDEN/64)/8, tb, 0, stream>>>(
        e_bf, EMBEDN, w_obs_t, EMBEDN, hobs_raw, HIDDEN,
        BT, HIDDEN, EMBEDN, nullptr);
    k_rmssilu<<<BT, tb, 0, stream>>>(hobs_raw, HIDDEN, hobs_bf, HIDDEN, b_obs, g_obs, HIDDEN);
    k_gemm_big<<<(BT/16)*(FLATSTOCH/64)/8, tb, 0, stream>>>(
        hobs_bf, HIDDEN, w_logit_t, HIDDEN, out_logits, FLATSTOCH,
        BT, FLATSTOCH, HIDDEN, b_logit);
    k_sample<<<(BT * STOCHN) / 256, tb, 0, stream>>>(
        out_logits, unif, reset, out_stochs, stoch_bf);

    // ---- Phase 2: sequential GRU recurrence ----
    for (int t = 0; t < LL; ++t) {
        // x0_raw = deterM @ w_in0      (32 x 1024, K=4096)
        k_gemm32<<<(HIDDEN/16)/8, tb, 0, stream>>>(
            deter_mbf, DETERN, w_in0_t, DETERN, xraw, 3*HIDDEN,
            HIDDEN, DETERN, nullptr, HIDDEN, 0);
        // x1_raw = stochM_{t} @ w_in1  (32 x 1024, K=1024)
        k_gemm32<<<(HIDDEN/16)/8, tb, 0, stream>>>(
            stoch_bf + (size_t)t * BB * FLATSTOCH, FLATSTOCH,
            w_in1_t, FLATSTOCH, xraw + HIDDEN, 3*HIDDEN,
            HIDDEN, FLATSTOCH, nullptr, HIDDEN, 0);
        // x2_raw = norm(mask(act)) @ w_in2   (K=6, VALU)
        k_act<<<(BB * HIDDEN) / 256, tb, 0, stream>>>(
            action, reset, w_in2, xraw + 2 * HIDDEN, t);
        // rms+silu for the three segments -> x_bf
        k_rmssilu_x<<<dim3(BB, 3), tb, 0, stream>>>(
            xraw, x_bf, b_in0, g_in0, b_in1, g_in1, b_in2, g_in2);
        // block-diagonal hidden matmul (32 x 4096, K=3584 per block)
        k_gemm_hid32<<<(DETERN/16)/8, tb, 0, stream>>>(deter_mbf, x_bf, w_hid_t, h_raw);
        // rms over full 4096 + silu -> h_bf
        k_rmssilu<<<BB, tb, 0, stream>>>(h_raw, DETERN, h_bf, DETERN, b_hid, g_hid, DETERN);
        // grouped GRU-gate matmul (32 x 12288, K=512 per block)
        k_gemm32<<<(3*DETERN/16)/8, tb, 0, stream>>>(
            h_bf, DETERN, w_gru_t, DPB, g3, 3*DETERN,
            3*DETERN, DPB, nullptr, 3*DPB, DPB);
        // gates + state update + masked bf16 deter for t+1 + deters output
        k_gru_update<<<(BB * DETERN) / 256, tb, 0, stream>>>(
            g3, b_gru, reset, deter_f32, deter_mbf, out_deters, t);
    }
}
